// VectorQuantizerLowRank_47923245089214
// MI455X (gfx1250) — matrix operand order repaired and checked by reading, not compiled, not run on hardware
//
#include <hip/hip_runtime.h>
#include <hip/hip_bf16.h>
#include <math.h>

typedef __attribute__((ext_vector_type(16))) __bf16 v16bf;
typedef __attribute__((ext_vector_type(8)))  float  v8f;
typedef unsigned short u16;
typedef unsigned int   u32;

#define N_B 8192
#define N_K 2048
#define N_D 768
#define SK_EPS 0.01f
#define BETA_C 0.25f

__device__ __forceinline__ u16 f32_bf16(float f) {
  u32 u = __float_as_uint(f);
  u32 r = u + 0x7FFFu + ((u >> 16) & 1u);   // round-to-nearest-even
  return (u16)(r >> 16);
}
__device__ __forceinline__ float bf16_f32(u16 h) {
  return __uint_as_float(((u32)h) << 16);
}

// ---- emb = lora_a @ lora_b : [2048,32]@[32,768] ----
__global__ void k_emb(const float* __restrict__ a, const float* __restrict__ b,
                      float* __restrict__ em) {
  int e = blockIdx.x * 256 + threadIdx.x;
  int n = e / N_D, dd = e % N_D;
  float s = 0.f;
#pragma unroll
  for (int r = 0; r < 32; ++r) s += a[n * 32 + r] * b[r * N_D + dd];
  em[e] = s;
}

// ---- split rows into bf16 hi/lo + row squared-norms ----
__global__ void k_prep(const float* __restrict__ src, u16* __restrict__ hi,
                       u16* __restrict__ lo, float* __restrict__ sq) {
  __shared__ float sd[256];
  int row = blockIdx.x, t = threadIdx.x;
  float acc = 0.f;
  for (int c = t; c < N_D; c += 256) {
    float v = src[(size_t)row * N_D + c];
    acc += v * v;
    u16 h = f32_bf16(v);
    float hv = bf16_f32(h);
    hi[(size_t)row * N_D + c] = h;
    lo[(size_t)row * N_D + c] = f32_bf16(v - hv);
  }
  sd[t] = acc; __syncthreads();
  for (int s = 128; s > 0; s >>= 1) { if (t < s) sd[t] += sd[t + s]; __syncthreads(); }
  if (t == 0) sq[row] = sd[0];
}

// ---- g = X @ Eᵀ via bf16-split WMMA (3 wmma per 32-wide K step) ----
__global__ void __launch_bounds__(256) k_gemm(const u16* __restrict__ Xhi,
                                              const u16* __restrict__ Xlo,
                                              const u16* __restrict__ Ehi,
                                              const u16* __restrict__ Elo,
                                              float* __restrict__ g) {
  int wave = threadIdx.x >> 5, lane = threadIdx.x & 31;
  int tileM = blockIdx.x * 16;
  int tileN = blockIdx.y * 128 + wave * 16;
  int aRow = tileM + (lane & 15);
  int aOff = (lane < 16) ? 0 : 8;     // A 16x32 bf16 lane striping (ISA 7.12.2)
  int bRow = tileN + (lane & 15);     // B column N = E row
  int bOff = (lane < 16) ? 0 : 8;     // B 32x16 bf16: same K striping as A
  v8f acc = {0.f, 0.f, 0.f, 0.f, 0.f, 0.f, 0.f, 0.f};
  const u16* pxh = Xhi + (size_t)aRow * N_D + aOff;
  const u16* pxl = Xlo + (size_t)aRow * N_D + aOff;
  const u16* peh = Ehi + (size_t)bRow * N_D + bOff;
  const u16* pel = Elo + (size_t)bRow * N_D + bOff;
  for (int k0 = 0; k0 < N_D; k0 += 32) {
    union { uint4 q[2]; v16bf v; } ah, al, bh, bl;
    ah.q[0] = *(const uint4*)(pxh + k0);  ah.q[1] = *(const uint4*)(pxh + k0 + 16);
    al.q[0] = *(const uint4*)(pxl + k0);  al.q[1] = *(const uint4*)(pxl + k0 + 16);
    bh.q[0] = *(const uint4*)(peh + k0);  bh.q[1] = *(const uint4*)(peh + k0 + 16);
    bl.q[0] = *(const uint4*)(pel + k0);  bl.q[1] = *(const uint4*)(pel + k0 + 16);
    acc = __builtin_amdgcn_wmma_f32_16x16x32_bf16(false, ah.v, false, bh.v, (short)0, acc, false, false);
    acc = __builtin_amdgcn_wmma_f32_16x16x32_bf16(false, ah.v, false, bl.v, (short)0, acc, false, false);
    acc = __builtin_amdgcn_wmma_f32_16x16x32_bf16(false, al.v, false, bh.v, (short)0, acc, false, false);
  }
  int col = tileN + (lane & 15);
  int rbase = tileM + ((lane < 16) ? 0 : 8);
#pragma unroll
  for (int r = 0; r < 8; ++r)
    g[(size_t)(rbase + r) * N_K + col] = acc[r];
}

// ---- per-sample-row min/max of d ----
__global__ void k_rowstat(const float* __restrict__ g, const float* __restrict__ xsq,
                          const float* __restrict__ esq, float* __restrict__ rowmin,
                          float* __restrict__ rowmax) {
  __shared__ float smn[256], smx[256];
  int i = blockIdx.x, t = threadIdx.x;
  float xs = xsq[i];
  const float4* gp = (const float4*)(g + (size_t)i * N_K + t * 8);
  const float4* ep = (const float4*)(esq + t * 8);
  float gv[8], ev[8];
  *(float4*)gv = gp[0]; *(float4*)(gv + 4) = gp[1];
  *(float4*)ev = ep[0]; *(float4*)(ev + 4) = ep[1];
  float mn = 3.4e38f, mx = -3.4e38f;
#pragma unroll
  for (int q = 0; q < 8; ++q) {
    float d = xs + ev[q] - 2.f * gv[q];
    mn = fminf(mn, d); mx = fmaxf(mx, d);
  }
  smn[t] = mn; smx[t] = mx; __syncthreads();
  for (int s = 128; s > 0; s >>= 1) {
    if (t < s) { smn[t] = fminf(smn[t], smn[t + s]); smx[t] = fmaxf(smx[t], smx[t + s]); }
    __syncthreads();
  }
  if (t == 0) { rowmin[i] = smn[0]; rowmax[i] = smx[0]; }
}

__global__ void k_globstat(const float* __restrict__ rowmin, const float* __restrict__ rowmax,
                           float* __restrict__ sc) {
  __shared__ float smn[256], smx[256];
  int t = threadIdx.x;
  float mn = 3.4e38f, mx = -3.4e38f;
  for (int i = t; i < N_B; i += 256) { mn = fminf(mn, rowmin[i]); mx = fmaxf(mx, rowmax[i]); }
  smn[t] = mn; smx[t] = mx; __syncthreads();
  for (int s = 128; s > 0; s >>= 1) {
    if (t < s) { smn[t] = fminf(smn[t], smn[t + s]); smx[t] = fmaxf(smx[t], smx[t + s]); }
    __syncthreads();
  }
  if (t == 0) {
    float middle = 0.5f * (smx[0] + smn[0]);
    float amp = smx[0] - middle + 1e-5f;
    sc[0] = middle; sc[1] = 1.f / amp;
  }
}

// ---- dc (f64 out) + row-stabilized kernel matrix M in bf16 ----
__global__ void k_makeM(const float* __restrict__ g, const float* __restrict__ xsq,
                        const float* __restrict__ esq, const float* __restrict__ rowmin,
                        const float* __restrict__ sc, u16* __restrict__ Mb,
                        u32* __restrict__ out_dc) {
  size_t e = (size_t)blockIdx.x * 256 + threadIdx.x;
  int i = (int)(e >> 11);
  int j = (int)(e & 2047);
  float d = xsq[i] + esq[j] - 2.f * g[e];
  float middle = sc[0], inv_amp = sc[1];
  float dc = (d - middle) * inv_amp;
  union { double dv; u32 u[2]; } cv; cv.dv = (double)dc;
  out_dc[2 * e] = cv.u[0]; out_dc[2 * e + 1] = cv.u[1];   // unaligned-safe f64 store
  float z = -(d - rowmin[i]) * inv_amp * (1.0f / SK_EPS); // <= 0, no overflow
  Mb[e] = f32_bf16(expf(z));
}

__global__ void k_init(float* __restrict__ alpha, float* __restrict__ lossacc) {
  int i = blockIdx.x * 256 + threadIdx.x;
  if (i < N_B) alpha[i] = 1.f;
  if (i == 0) lossacc[0] = 0.f;
}

// ---- Sinkhorn: column sums (over samples) -> partials ----
__global__ void k_colsum(const u16* __restrict__ Mb, const float* __restrict__ alpha,
                         float* __restrict__ partial) {
  int blk = blockIdx.x, t = threadIdx.x;
  int r0 = blk * 256;
  float s[8] = {0, 0, 0, 0, 0, 0, 0, 0};
  for (int r = r0; r < r0 + 256; ++r) {
    float av = alpha[r];
    uint4 m = *(const uint4*)(Mb + (size_t)r * N_K + t * 8);
    const u16* mp = (const u16*)&m;
#pragma unroll
    for (int q = 0; q < 8; ++q) s[q] += bf16_f32(mp[q]) * av;
  }
#pragma unroll
  for (int q = 0; q < 8; ++q) partial[(size_t)blk * N_K + t * 8 + q] = s[q];
}

__global__ void k_beta(const float* __restrict__ partial, float* __restrict__ beta) {
  int j = blockIdx.x * 256 + threadIdx.x;
  float s = 0.f;
#pragma unroll
  for (int p = 0; p < 32; ++p) s += partial[(size_t)p * N_K + j];
  beta[j] = 1.f / ((float)N_K * fmaxf(s, 1e-30f));
}

// ---- Sinkhorn: row sums (over codes) -> alpha ----
__global__ void k_rowalpha(const u16* __restrict__ Mb, const float* __restrict__ beta,
                           float* __restrict__ alpha) {
  __shared__ float sd[256];
  int i = blockIdx.x, t = threadIdx.x;
  uint4 m = *(const uint4*)(Mb + (size_t)i * N_K + t * 8);
  const u16* mp = (const u16*)&m;
  float bv[8];
  *(float4*)bv = *(const float4*)(beta + t * 8);
  *(float4*)(bv + 4) = *(const float4*)(beta + t * 8 + 4);
  float s = 0.f;
#pragma unroll
  for (int q = 0; q < 8; ++q) s += bf16_f32(mp[q]) * bv[q];
  sd[t] = s; __syncthreads();
  for (int k = 128; k > 0; k >>= 1) { if (t < k) sd[t] += sd[t + k]; __syncthreads(); }
  if (t == 0) alpha[i] = 1.f / ((float)N_B * fmaxf(sd[0], 1e-30f));
}

// ---- argmax, gather x_q, straight-through out, loss accumulate ----
__global__ void k_argmax(const u16* __restrict__ Mb, const float* __restrict__ beta,
                         const float* __restrict__ em, const float* __restrict__ x,
                         float* __restrict__ out_xq, u32* __restrict__ out_idx,
                         float* __restrict__ lossacc) {
  __shared__ float sv[256];
  __shared__ int si[256];
  __shared__ int sidx;
  int i = blockIdx.x, t = threadIdx.x;
  uint4 m = *(const uint4*)(Mb + (size_t)i * N_K + t * 8);
  const u16* mp = (const u16*)&m;
  float best = -1.f; int bj = t * 8;
#pragma unroll
  for (int q = 0; q < 8; ++q) {
    int j = t * 8 + q;
    float v = bf16_f32(mp[q]) * beta[j];
    if (v > best) { best = v; bj = j; }
  }
  sv[t] = best; si[t] = bj; __syncthreads();
  for (int k = 128; k > 0; k >>= 1) {
    if (t < k) {
      if (sv[t + k] > sv[t] || (sv[t + k] == sv[t] && si[t + k] < si[t])) {
        sv[t] = sv[t + k]; si[t] = si[t + k];
      }
    }
    __syncthreads();
  }
  if (t == 0) { sidx = si[0]; out_idx[2 * i] = (u32)si[0]; out_idx[2 * i + 1] = 0u; }
  __syncthreads();
  int idx = sidx;
  float acc = 0.f;
  for (int c = t; c < N_D; c += 256) {
    float xq = em[(size_t)idx * N_D + c];
    float xi = x[(size_t)i * N_D + c];
    float diff = xq - xi;
    out_xq[(size_t)i * N_D + c] = xi + diff;   // straight-through, as reference computes it
    acc += diff * diff;
  }
  sv[t] = acc; __syncthreads();
  for (int k = 128; k > 0; k >>= 1) { if (t < k) sv[t] += sv[t + k]; __syncthreads(); }
  if (t == 0) atomicAdd(lossacc, sv[0]);
}

__global__ void k_loss(const float* __restrict__ lossacc, float* __restrict__ out_loss) {
  out_loss[0] = (1.f + BETA_C) * lossacc[0] / (float)((size_t)N_B * N_D);
}

extern "C" void kernel_launch(void* const* d_in, const int* in_sizes, int n_in,
                              void* d_out, int out_size, void* d_ws, size_t ws_size,
                              hipStream_t stream) {
  const float* x  = (const float*)d_in[0];
  const float* la = (const float*)d_in[1];
  const float* lb = (const float*)d_in[2];

  char* w = (char*)d_ws;
  float* g       = (float*)w;  w += (size_t)N_B * N_K * 4;
  u16*   Mb      = (u16*)w;    w += (size_t)N_B * N_K * 2;
  u16*   xhi     = (u16*)w;    w += (size_t)N_B * N_D * 2;
  u16*   xlo     = (u16*)w;    w += (size_t)N_B * N_D * 2;
  u16*   ehi     = (u16*)w;    w += (size_t)N_K * N_D * 2;
  u16*   elo     = (u16*)w;    w += (size_t)N_K * N_D * 2;
  float* em      = (float*)w;  w += (size_t)N_K * N_D * 4;
  float* xsq     = (float*)w;  w += (size_t)N_B * 4;
  float* esq     = (float*)w;  w += (size_t)N_K * 4;
  float* rowmin  = (float*)w;  w += (size_t)N_B * 4;
  float* rowmax  = (float*)w;  w += (size_t)N_B * 4;
  float* alpha   = (float*)w;  w += (size_t)N_B * 4;
  float* beta    = (float*)w;  w += (size_t)N_K * 4;
  float* partial = (float*)w;  w += (size_t)32 * N_K * 4;
  float* sc      = (float*)w;  w += 64;
  float* lossacc = (float*)w;  w += 64;

  char* ob = (char*)d_out;
  float* out_xq   = (float*)ob;
  float* out_loss = (float*)(ob + (size_t)N_B * N_D * 4);
  u32*   out_idx  = (u32*)(ob + (size_t)N_B * N_D * 4 + 4);
  u32*   out_dc   = (u32*)(ob + (size_t)N_B * N_D * 4 + 4 + (size_t)N_B * 8);

  k_emb<<<(N_K * N_D) / 256, 256, 0, stream>>>(la, lb, em);
  k_prep<<<N_K, 256, 0, stream>>>(em, ehi, elo, esq);
  k_prep<<<N_B, 256, 0, stream>>>(x, xhi, xlo, xsq);
  k_gemm<<<dim3(N_B / 16, N_K / 128), 256, 0, stream>>>(xhi, xlo, ehi, elo, g);
  k_rowstat<<<N_B, 256, 0, stream>>>(g, xsq, esq, rowmin, rowmax);
  k_globstat<<<1, 256, 0, stream>>>(rowmin, rowmax, sc);
  k_makeM<<<(int)(((size_t)N_B * N_K) / 256), 256, 0, stream>>>(g, xsq, esq, rowmin, sc, Mb, out_dc);
  k_init<<<N_B / 256, 256, 0, stream>>>(alpha, lossacc);
  for (int it = 0; it < 100; ++it) {
    k_colsum<<<32, 256, 0, stream>>>(Mb, alpha, partial);
    k_beta<<<N_K / 256, 256, 0, stream>>>(partial, beta);
    k_rowalpha<<<N_B, 256, 0, stream>>>(Mb, beta, alpha);
  }
  k_argmax<<<N_B, 256, 0, stream>>>(Mb, beta, em, x, out_xq, out_idx, lossacc);
  k_loss<<<1, 1, 0, stream>>>(lossacc, out_loss);
}
